// spixel_upsample2_5d_78752520339849
// MI455X (gfx1250) — matrix operand order repaired and checked by reading, not compiled, not run on hardware
//
#include <hip/hip_runtime.h>

typedef __attribute__((ext_vector_type(16))) _Float16 v16h;
typedef __attribute__((ext_vector_type(8)))  _Float16 v8h;
typedef __attribute__((ext_vector_type(8)))  float    v8f;
typedef __attribute__((ext_vector_type(4)))  float    v4f;
typedef __attribute__((ext_vector_type(4)))  int      v4i;

#define EPSV  1e-5f
#define SLOPE 0.01f
#define AS1 __attribute__((address_space(1)))
#define AS3 __attribute__((address_space(3)))

#if __has_builtin(__builtin_amdgcn_global_load_async_to_lds_b128) && \
    __has_builtin(__builtin_amdgcn_s_wait_asynccnt)
#define HAVE_ASYNC_LDS 1
#else
#define HAVE_ASYNC_LDS 0
#endif

__device__ __forceinline__ float lrelu(float y) { return y > 0.f ? y : SLOPE * y; }

// A-fragment (16x32 f16) k index for half-element i in lane-group g (contiguous: k = 16g + i)
__device__ __forceinline__ int a_k_of(int g, int i) {
    return (g << 4) + i;
}

// Weight-fragment table offsets (in halfs) inside wf[]
#define WF_K1 0        // 6 variants (mt,t)       x 32 lanes x 16 = 3072
#define WF_K2 3072     // 16 variants (td,th,tw)  x 32 lanes x 16 = 8192
#define WF_K3 11264    // 3 variants (t)          x 32 lanes x 16 = 1536
#define WF_K4 12800    // 16 variants (td,th,tw)  x 16 ic          = 256
#define WF_TOT 13056

// ---------------------------------------------------------------------------
// kw: pre-pack all WMMA A-fragments (and k4 weight vectors) as f16 in
// fragment order, so GEMM waves fetch one 32B v16h per fragment.
// ---------------------------------------------------------------------------
__global__ __launch_bounds__(256)
void kw_prep(const float* __restrict__ w0a, const float* __restrict__ w0b,
             const float* __restrict__ w1a, const float* __restrict__ w1b,
             _Float16* __restrict__ wf)
{
    const int tid = threadIdx.x;
    for (int e = tid; e < 3072; e += 256) {           // k1: wc[oc,ic,t]=w0a[ic,oc,2-t]
        int i = e & 15, lane = (e >> 4) & 31, v = e >> 9;   // v = mt*3+t
        int mt = v / 3, t = v % 3;
        int g = lane >> 4, ln = lane & 15;
        int ic = a_k_of(g, i), oc = mt * 16 + ln;
        wf[WF_K1 + e] = (_Float16)w0a[(ic * 32 + oc) * 3 + (2 - t)];
    }
    for (int e = tid; e < 8192; e += 256) {           // k2: w0b[ic,oc,td,th,tw]
        int i = e & 15, lane = (e >> 4) & 31, v = e >> 9;   // v = (td*2+th)*2+tw
        int tw = v & 1, th = (v >> 1) & 1, td = v >> 2;
        int g = lane >> 4, ln = lane & 15;
        int ic = a_k_of(g, i);
        wf[WF_K2 + e] = (_Float16)w0b[(((ic * 16 + ln) * 4 + td) * 2 + th) * 2 + tw];
    }
    for (int e = tid; e < 1536; e += 256) {           // k3: wc[oc,ic,t]=w1a[ic,oc,2-t]
        int i = e & 15, lane = (e >> 4) & 31, t = e >> 9;
        int g = lane >> 4, ln = lane & 15;
        int ic = a_k_of(g, i);
        wf[WF_K3 + e] = (_Float16)w1a[(ic * 16 + ln) * 3 + (2 - t)];
    }
    for (int e = tid; e < 256; e += 256) {            // k4: w1b[ic,0,td,th,tw]
        int ic = e & 15, v = e >> 4;
        int tw = v & 1, th = (v >> 1) & 1, td = v >> 2;
        wf[WF_K4 + e] = (_Float16)w1b[((ic * 4 + td) * 2 + th) * 2 + tw];
    }
}

// ---------------------------------------------------------------------------
// k0a: cv0 (16,48,128,256) f32 planar -> f16 channel-last (50MB, L2-resident)
// ---------------------------------------------------------------------------
__global__ __launch_bounds__(256)
void k0_cvt_cv0(const float* __restrict__ cv0, _Float16* __restrict__ h0)
{
    const int p = blockIdx.x * 256 + threadIdx.x;   // 1,572,864 positions
    v16h v;
    #pragma unroll
    for (int c = 0; c < 16; ++c) v[c] = (_Float16)cv0[c * 1572864 + p];
    *(v16h*)(h0 + (size_t)p * 16) = v;
}

// ---------------------------------------------------------------------------
// k0b: cv1 (32,24,64,128) f32 planar -> f16 channel-last (12.6MB)
// ---------------------------------------------------------------------------
__global__ __launch_bounds__(256)
void k0_cvt_cv1(const float* __restrict__ cv1, _Float16* __restrict__ cv1h)
{
    const int p = blockIdx.x * 256 + threadIdx.x;   // 196,608 positions
    v16h a, b;
    #pragma unroll
    for (int c = 0; c < 16; ++c) {
        a[c] = (_Float16)cv1[c * 196608 + p];
        b[c] = (_Float16)cv1[(16 + c) * 196608 + p];
    }
    *(v16h*)(cv1h + (size_t)p * 32)      = a;
    *(v16h*)(cv1h + (size_t)p * 32 + 16) = b;
}

// ---------------------------------------------------------------------------
// L1: deconv3d(cv1, w0a, s1, pad(1,0,0)) + bn_lrelu -> x1h f16 [d][s][ic32]
// M=32 (two 16-tiles), K = 3 taps x 32 ic, N = 16 spatial per wave.
// A and B fragments are each one 32B contiguous load.
// ---------------------------------------------------------------------------
__global__ __launch_bounds__(256)
void k1_deconv_a(const _Float16* __restrict__ cv1h, const _Float16* __restrict__ wf,
                 const float* __restrict__ g0, const float* __restrict__ b0,
                 _Float16* __restrict__ x1h)
{
    const int lane = threadIdx.x & 31;
    const int tile = blockIdx.x * 8 + (threadIdx.x >> 5);   // 12288 wave tiles
    const int d  = tile >> 9;                               // 0..23
    const int s0 = (tile & 511) << 4;
    const int g  = lane >> 4, ln = lane & 15;

    v8f c0 = {}, c1 = {};
    #pragma unroll
    for (int t = 0; t < 3; ++t) {
        const int dz = d + t - 1;
        v16h B;
        if (dz >= 0 && dz < 24) {                 // wave-uniform
            B = *(const v16h*)(cv1h + (size_t)(dz * 8192 + s0 + ln) * 32 + 16 * g);
        } else {
            #pragma unroll
            for (int i = 0; i < 16; ++i) B[i] = (_Float16)0.f;
        }
        const v16h A0 = *(const v16h*)(wf + WF_K1 + ((0 * 3 + t) * 32 + lane) * 16);
        const v16h A1 = *(const v16h*)(wf + WF_K1 + ((1 * 3 + t) * 32 + lane) * 16);
        c0 = __builtin_amdgcn_wmma_f32_16x16x32_f16(false, A0, false, B, (short)0, c0, false, false);
        c1 = __builtin_amdgcn_wmma_f32_16x16x32_f16(false, A1, false, B, (short)0, c1, false, false);
    }

    const float inv = rsqrtf(1.f + EPSV);
    const int p  = d * 8192 + s0 + ln;
    const int cb = 8 * g;                         // D rows m = v + 8g
    const v4f ga = *(const v4f*)(g0 + cb),      gb = *(const v4f*)(g0 + cb + 4);
    const v4f ba = *(const v4f*)(b0 + cb),      bb = *(const v4f*)(b0 + cb + 4);
    const v4f gc = *(const v4f*)(g0 + 16 + cb), gd = *(const v4f*)(g0 + 16 + cb + 4);
    const v4f bc = *(const v4f*)(b0 + 16 + cb), bd = *(const v4f*)(b0 + 16 + cb + 4);
    v8h o0, o1;
    #pragma unroll
    for (int v = 0; v < 4; ++v) {
        o0[v]     = (_Float16)lrelu(c0[v]     * (ga[v] * inv) + ba[v]);
        o0[v + 4] = (_Float16)lrelu(c0[v + 4] * (gb[v] * inv) + bb[v]);
        o1[v]     = (_Float16)lrelu(c1[v]     * (gc[v] * inv) + bc[v]);
        o1[v + 4] = (_Float16)lrelu(c1[v + 4] * (gd[v] * inv) + bd[v]);
    }
    *(v8h*)(x1h + (size_t)p * 32 + cb)      = o0;   // ch 8g..8g+7
    *(v8h*)(x1h + (size_t)p * 32 + 16 + cb) = o1;   // ch 16+8g..
}

// ---------------------------------------------------------------------------
// L2: deconv3d(x1, w0b, s2, pad(1,0,0)) + bn_lrelu -> x2h f16 [d][s][oc16]
// Per parity: 2 depth taps -> M=16, K = 2 x 32 ic, N = 16 wi per wave
// ---------------------------------------------------------------------------
__global__ __launch_bounds__(256)
void k2_deconv_b(const _Float16* __restrict__ x1h, const _Float16* __restrict__ wf,
                 const float* __restrict__ g1, const float* __restrict__ b1,
                 _Float16* __restrict__ x2h)
{
    const int lane = threadIdx.x & 31;
    const int tile = blockIdx.x * 8 + (threadIdx.x >> 5);   // 98304 wave tiles
    const int sub  = tile & 15;
    const int row  = tile >> 4;
    const int dO   = row >> 7;          // 0..47
    const int ho   = row & 127;
    const int tw   = sub & 1;
    const int wi0  = (sub >> 1) << 4;
    const int g = lane >> 4, ln = lane & 15;

    int td[2], di[2];
    if ((dO & 1) == 0) { td[0] = 1; td[1] = 3; di[0] = dO >> 1;       di[1] = (dO >> 1) - 1; }
    else               { td[0] = 0; td[1] = 2; di[0] = (dO + 1) >> 1; di[1] = (dO - 1) >> 1; }
    const int th = ho & 1, hi = ho >> 1;

    v8f c = {};
    #pragma unroll
    for (int j = 0; j < 2; ++j) {
        const int vv = (td[j] * 2 + th) * 2 + tw;
        const v16h A = *(const v16h*)(wf + WF_K2 + ((size_t)vv * 32 + lane) * 16);
        v16h B;
        const int dv = di[j];
        if (dv >= 0 && dv < 24) {                 // wave-uniform
            const size_t p = (size_t)(dv * 8192 + hi * 128 + wi0 + ln);
            B = *(const v16h*)(x1h + p * 32 + 16 * g);   // ic = g*16 .. g*16+15
        } else {
            #pragma unroll
            for (int i = 0; i < 16; ++i) B[i] = (_Float16)0.f;
        }
        c = __builtin_amdgcn_wmma_f32_16x16x32_f16(false, A, false, B, (short)0, c, false, false);
    }

    const float inv = rsqrtf(1.f + EPSV);
    const int wo = ((wi0 + ln) << 1) + tw;
    const size_t p = (size_t)dO * 32768 + ho * 256 + wo;
    const int cb = 8 * g;
    const v4f ga = *(const v4f*)(g1 + cb), gb = *(const v4f*)(g1 + cb + 4);
    const v4f ba = *(const v4f*)(b1 + cb), bb = *(const v4f*)(b1 + cb + 4);
    v8h o;
    #pragma unroll
    for (int v = 0; v < 4; ++v) {
        o[v]     = (_Float16)lrelu(c[v]     * (ga[v] * inv) + ba[v]);
        o[v + 4] = (_Float16)lrelu(c[v + 4] * (gb[v] * inv) + bb[v]);
    }
    *(v8h*)(x2h + p * 16 + cb) = o;
}

// ---------------------------------------------------------------------------
// L3: deconv3d(concat(x2,cv0f16), w1a, s1, pad(1,0,0)) + bn_lrelu -> x3h
// Virtual concat: lane-group 0 loads 32B from x2h, group 1 from h0 (L2-hot)
// ---------------------------------------------------------------------------
__global__ __launch_bounds__(256)
void k3_conv(const _Float16* __restrict__ x2h, const _Float16* __restrict__ h0,
             const _Float16* __restrict__ wf, const float* __restrict__ g2,
             const float* __restrict__ b2, _Float16* __restrict__ x3h)
{
    const int lane = threadIdx.x & 31;
    const int tile = blockIdx.x * 8 + (threadIdx.x >> 5);   // 98304 wave tiles
    const int dz = tile >> 11;                              // 0..47
    const int s0 = (tile & 2047) << 4;
    const int g = lane >> 4, ln = lane & 15;

    v8f c = {};
    #pragma unroll
    for (int t = 0; t < 3; ++t) {
        const int dv = dz + t - 1;
        const v16h A = *(const v16h*)(wf + WF_K3 + ((size_t)t * 32 + lane) * 16);
        v16h B;
        if (dv >= 0 && dv < 48) {                 // wave-uniform
            const size_t p = (size_t)dv * 32768 + s0 + ln;
            const _Float16* src = (g == 0) ? (x2h + p * 16) : (h0 + p * 16);
            B = *(const v16h*)src;                // 32B: ic g*16 .. g*16+15
        } else {
            #pragma unroll
            for (int i = 0; i < 16; ++i) B[i] = (_Float16)0.f;
        }
        c = __builtin_amdgcn_wmma_f32_16x16x32_f16(false, A, false, B, (short)0, c, false, false);
    }

    const float inv = rsqrtf(1.f + EPSV);
    const size_t p = (size_t)dz * 32768 + s0 + ln;
    const int cb = 8 * g;
    const v4f ga = *(const v4f*)(g2 + cb), gb = *(const v4f*)(g2 + cb + 4);
    const v4f ba = *(const v4f*)(b2 + cb), bb = *(const v4f*)(b2 + cb + 4);
    v8h o;
    #pragma unroll
    for (int v = 0; v < 4; ++v) {
        o[v]     = (_Float16)lrelu(c[v]     * (ga[v] * inv) + ba[v]);
        o[v + 4] = (_Float16)lrelu(c[v + 4] * (gb[v] * inv) + bb[v]);
    }
    *(v8h*)(x3h + p * 16 + cb) = o;
}

// ---------------------------------------------------------------------------
// L4: deconv3d(x3, w1b, s2, pad(1,0,0)) + bn_lrelu -> cost f32 (96,256,512)
// M=1 -> VALU. Each tap: one 32B activation load + one 32B weight load.
// ---------------------------------------------------------------------------
__global__ __launch_bounds__(256)
void k4_deconv_d(const _Float16* __restrict__ x3h, const _Float16* __restrict__ wf,
                 const float* __restrict__ g3, const float* __restrict__ b3,
                 float* __restrict__ cost)
{
    const int idx = blockIdx.x * 256 + threadIdx.x;  // 12,582,912
    const int wo = idx & 511;
    const int t  = idx >> 9;
    const int ho = t & 255;
    const int dO = t >> 8;                            // 0..95
    int td[2], di[2];
    if ((dO & 1) == 0) { td[0] = 1; td[1] = 3; di[0] = dO >> 1;       di[1] = (dO >> 1) - 1; }
    else               { td[0] = 0; td[1] = 2; di[0] = (dO + 1) >> 1; di[1] = (dO - 1) >> 1; }
    const int th = ho & 1, hi = ho >> 1;
    const int tw = wo & 1, wi = wo >> 1;

    float acc = 0.f;
    #pragma unroll
    for (int j = 0; j < 2; ++j) {
        const int dv = di[j];
        if (dv < 0 || dv >= 48) continue;
        const v16h s  = *(const v16h*)(x3h + ((size_t)dv * 32768 + hi * 256 + wi) * 16);
        const v16h wv = *(const v16h*)(wf + WF_K4 + ((td[j] * 2 + th) * 2 + tw) * 16);
        #pragma unroll
        for (int ic = 0; ic < 16; ++ic)
            acc = fmaf((float)wv[ic], (float)s[ic], acc);
    }
    float y = acc * (g3[0] * rsqrtf(1.f + EPSV)) + b3[0];
    cost[idx] = lrelu(y);
}

// ---------------------------------------------------------------------------
// Aggregation: agg[d,h,w] = sum_k sp[k,h,w] * cost[d, h-4hy, w-4wx]
// Block = one (h, 256-wide w strip). 3 rows x 264 cols staged per depth via
// async global->LDS b128 (double buffered over 96 depths), 9-tap from LDS.
// ---------------------------------------------------------------------------
__global__ __launch_bounds__(256)
void k5_agg(const float* __restrict__ cost, const float* __restrict__ sp,
            float* __restrict__ out)
{
    __shared__ float smem[2][3][272];
    const int tid = threadIdx.x;
    const int h   = blockIdx.x >> 1;
    const int w0  = (blockIdx.x & 1) << 8;
    const int w   = w0 + tid;
    const int hw  = h * 512 + w;

    const int hy[9] = {1, 1, 1, 0, 0, 0, -1, -1, -1};
    const int wx[9] = {1, 0, -1, 1, 0, -1, 1, 0, -1};
    float wgt[9]; bool ok[9];
    #pragma unroll
    for (int k = 0; k < 9; ++k) {
        const int sh = h - 4 * hy[k], sw = w - 4 * wx[k];
        ok[k]  = (sh >= 0) && (sh < 256) && (sw >= 0) && (sw < 512);
        wgt[k] = ok[k] ? sp[k * 131072 + hw] : 0.f;
    }

    auto stage = [&](int d, int buf) {
#if HAVE_ASYNC_LDS
        if (tid < 198) {                              // 3 rows x 66 x b128
            const int r = tid / 66, c = (tid - r * 66) * 4;
            const int rh = h + 4 * (r - 1);
            if (rh >= 0 && rh < 256)
                __builtin_amdgcn_global_load_async_to_lds_b128(
                    (AS1 v4i*)(cost + (size_t)d * 131072 + rh * 512 + (w0 - 4) + c),
                    (AS3 v4i*)&smem[buf][r][c], 0, 0);
        }
#else
        for (int e = tid; e < 792; e += 256) {
            const int r = e / 264, c = e - r * 264;
            const int rh = h + 4 * (r - 1);
            if (rh >= 0 && rh < 256)
                smem[buf][r][c] = cost[(size_t)d * 131072 + rh * 512 + (w0 - 4) + c];
        }
#endif
    };

    stage(0, 0);
    for (int d = 0; d < 96; ++d) {
        const int cur = d & 1;
#if HAVE_ASYNC_LDS
        __builtin_amdgcn_s_wait_asynccnt(0);
#endif
        __syncthreads();
        if (d + 1 < 96) stage(d + 1, cur ^ 1);
        float acc = 0.f;
        #pragma unroll
        for (int k = 0; k < 9; ++k)
            if (ok[k])
                acc = fmaf(wgt[k], smem[cur][1 - hy[k]][tid + 4 - 4 * wx[k]], acc);
        out[(size_t)d * 131072 + hw] = acc;
    }
}

// ---------------------------------------------------------------------------
// Launch.  Inputs: 0 cv1, 1 cv0, 2 sp, 3 w0a, 4 w0b, 5 w1a, 6 w1b,
//                  7 g0, 8 g1, 9 g2, 10 g3, 11 b0, 12 b1, 13 b2, 14 b3
// ---------------------------------------------------------------------------
extern "C" void kernel_launch(void* const* d_in, const int* in_sizes, int n_in,
                              void* d_out, int out_size, void* d_ws, size_t ws_size,
                              hipStream_t stream)
{
    const float* cv1 = (const float*)d_in[0];
    const float* cv0 = (const float*)d_in[1];
    const float* sp  = (const float*)d_in[2];
    const float* w0a = (const float*)d_in[3];
    const float* w0b = (const float*)d_in[4];
    const float* w1a = (const float*)d_in[5];
    const float* w1b = (const float*)d_in[6];
    const float* g0  = (const float*)d_in[7];
    const float* g1  = (const float*)d_in[8];
    const float* g2  = (const float*)d_in[9];
    const float* g3  = (const float*)d_in[10];
    const float* b0  = (const float*)d_in[11];
    const float* b1  = (const float*)d_in[12];
    const float* b2  = (const float*)d_in[13];
    const float* b3  = (const float*)d_in[14];

    float* W = (float*)d_ws;
    _Float16* h0   = (_Float16*)(W);              // 25,165,824 halfs (50MB)
    _Float16* x1h  = (_Float16*)(W + 12582912);   //  6,291,456 halfs (12.6MB)
    _Float16* x2h  = (_Float16*)(W + 15728640);   // 25,165,824 halfs (50MB)
    _Float16* x3h  = (_Float16*)(W + 28311552);   // 25,165,824 halfs (50MB)
    float*    cost = W + 40894464;                // 12,582,912 f32 (+64 pad)
    _Float16* cv1h = (_Float16*)(W + 53477440);   //  6,291,456 halfs (12.6MB)
    _Float16* wf   = (_Float16*)(W + 56623168);   // 13,056 halfs
    float*    out  = (float*)d_out;

    kw_prep    <<<1,     256, 0, stream>>>(w0a, w0b, w1a, w1b, wf);
    k0_cvt_cv0 <<<6144,  256, 0, stream>>>(cv0, h0);
    k0_cvt_cv1 <<<768,   256, 0, stream>>>(cv1, cv1h);
    k1_deconv_a<<<1536,  256, 0, stream>>>(cv1h, wf, g0, b0, x1h);
    k2_deconv_b<<<12288, 256, 0, stream>>>(x1h, wf, g1, b1, x2h);
    k3_conv    <<<12288, 256, 0, stream>>>(x2h, h0, wf, g2, b2, x3h);
    k4_deconv_d<<<49152, 256, 0, stream>>>(x3h, wf, g3, b3, cost);
    k5_agg     <<<512,   256, 0, stream>>>(cost, sp, out);
}
